// LocalTransformer_55551107006517
// MI455X (gfx1250) — matrix operand order repaired and checked by reading, not compiled, not run on hardware
//
#include <hip/hip_runtime.h>
#include <math.h>

typedef __attribute__((ext_vector_type(16))) __bf16 v16bf;
typedef __attribute__((ext_vector_type(8)))  float  v8f;

#define SEQL 8192
#define NBATCH 2
#define MTOK (SEQL*NBATCH)
#define DIMM 512
#define NHEADS 8
#define DHEAD 64
#define WINL 128
#define NWIN (SEQL/WINL)
#define CINNER 1024
#define KSZ 31

__device__ __forceinline__ float sigmf(float x){ return 1.f/(1.f+__expf(-x)); }

// -------------------- embedding gather --------------------
__global__ void embed_kernel(const int* __restrict__ tok, const float* __restrict__ te,
                             const float* __restrict__ pe, float* __restrict__ X){
  size_t n = (size_t)MTOK*DIMM;
  for(size_t i = (size_t)blockIdx.x*blockDim.x + threadIdx.x; i < n;
      i += (size_t)gridDim.x*blockDim.x){
    int m = (int)(i >> 9), d = (int)(i & 511);
    int t = m & (SEQL-1);
    X[i] = te[(size_t)tok[m]*DIMM + d] + pe[(size_t)t*DIMM + d];
  }
}

// -------------------- fp32 -> bf16 (no transpose) --------------------
__global__ void cvt_kernel(const float* __restrict__ in, __bf16* __restrict__ out, size_t n){
  for(size_t i = (size_t)blockIdx.x*blockDim.x + threadIdx.x; i < n;
      i += (size_t)gridDim.x*blockDim.x)
    out[i] = (__bf16)in[i];
}

// ---------- fp32 [K,N] -> bf16 transposed [N,K] (tiled, coalesced) ----------
__global__ __launch_bounds__(256) void cvt_t_kernel(const float* __restrict__ in,
    __bf16* __restrict__ out, int K, int N){
  __shared__ float tile[32][33];
  int n0 = blockIdx.x*32, k0 = blockIdx.y*32;
  int tx = threadIdx.x & 31, ty0 = threadIdx.x >> 5;
  #pragma unroll
  for(int r=0;r<4;r++){
    int ty = ty0 + r*8;
    tile[ty][tx] = in[(size_t)(k0+ty)*N + n0+tx];
  }
  __syncthreads();
  #pragma unroll
  for(int r=0;r<4;r++){
    int ty = ty0 + r*8;
    out[(size_t)(n0+ty)*K + k0+tx] = (__bf16)tile[tx][ty];
  }
}

// -------------------- layernorm over DIMM, bf16 output (1 wave = 1 token) ----
__global__ __launch_bounds__(256) void ln_kernel(const float* __restrict__ x,
    const float* __restrict__ g, const float* __restrict__ b, __bf16* __restrict__ out){
  int w = threadIdx.x >> 5, lane = threadIdx.x & 31;
  int token = blockIdx.x*8 + w;
  const float* xr = x + (size_t)token*DIMM;
  float v[16]; float s = 0.f, s2 = 0.f;
  #pragma unroll
  for(int j=0;j<16;j++){ float t = xr[lane*16+j]; v[j]=t; s+=t; s2+=t*t; }
  #pragma unroll
  for(int off=16; off>=1; off>>=1){ s += __shfl_xor(s, off, 32); s2 += __shfl_xor(s2, off, 32); }
  float mean = s*(1.f/DIMM);
  float var  = s2*(1.f/DIMM) - mean*mean;
  float rs = rsqrtf(var + 1e-5f);
  __bf16* orow = out + (size_t)token*DIMM;
  #pragma unroll
  for(int j=0;j<16;j++){ int c = lane*16+j; orow[c] = (__bf16)((v[j]-mean)*rs*g[c]+b[c]); }
}

// -------------------- dynamic position bias MLP (single block) --------------
__global__ __launch_bounds__(256) void posbias_kernel(
    const float* __restrict__ w1, const float* __restrict__ b1,
    const float* __restrict__ w2, const float* __restrict__ b2,
    const float* __restrict__ w3, const float* __restrict__ b3,
    float* __restrict__ bias){
  __shared__ float bj[256][8];
  int j = threadIdx.x;
  float h1[128], h2[128];
  for(int c=0;c<128;c++){ float s = (float)j*w1[c] + b1[c]; h1[c] = s*sigmf(s); }
  for(int c=0;c<128;c++){
    float s = b2[c];
    for(int cc=0;cc<128;cc++) s += h1[cc]*w2[cc*128+c];
    h2[c] = s*sigmf(s);
  }
  for(int hh=0;hh<8;hh++){
    float s = b3[hh];
    for(int cc=0;cc<128;cc++) s += h2[cc]*w3[cc*8+hh];
    bj[j][hh] = s;
  }
  __syncthreads();
  for(int idx=threadIdx.x; idx < 8*128*256; idx += 256){
    int hh = idx >> 15; int rem = idx & 32767; int i = rem >> 8; int j2 = rem & 255;
    int delta = i + 128 - j2; if(delta < 0) delta = -delta;
    bias[idx] = bj[delta][hh];
  }
}

// ---------- generic bf16 WMMA GEMM: C = act(A@BT^T + bias) + res ----------
// A[M,K] bf16 row-major, BT[N,K] bf16 row-major (pre-transposed weight),
// C[M,N] fp32. 128x128 block tile, 8 waves of 2x4 16x16 WMMA tiles, BK=32.
// All LDS fragment reads are K-contiguous -> ds_load_b128.
__global__ __launch_bounds__(256) void gemm_kernel(
    const __bf16* __restrict__ A, const __bf16* __restrict__ BT, float* __restrict__ C,
    const float* __restrict__ bias, const float* __restrict__ res,
    int M, int N, int K, int act){
  __shared__ __align__(16) __bf16 As[128][32];
  __shared__ __align__(16) __bf16 BsT[128][32];
  int tid = threadIdx.x;
  int w_id = tid >> 5, lane = tid & 31;
  int halfl = lane >> 4, lcol = lane & 15;
  int wm = w_id & 3, wn = w_id >> 2;      // 4 m-waves x 2 n-waves
  int m0 = blockIdx.y * 128, n0 = blockIdx.x * 128;
  v8f acc[2][4] = {};
  int trow = tid >> 1, tcolh = (tid & 1) << 4;   // 2 threads per 32-elem row
  for(int kt = 0; kt < K; kt += 32){
    const __bf16* ap = A  + (size_t)(m0 + trow)*K + kt + tcolh;
    const __bf16* bp = BT + (size_t)(n0 + trow)*K + kt + tcolh;
    *(uint4*)&As[trow][tcolh]   = *(const uint4*)ap;
    *(uint4*)&As[trow][tcolh+8] = *(const uint4*)(ap+8);
    *(uint4*)&BsT[trow][tcolh]   = *(const uint4*)bp;
    *(uint4*)&BsT[trow][tcolh+8] = *(const uint4*)(bp+8);
    if(kt + 32 < K){
      __builtin_prefetch(ap + 32, 0, 3);   // global_prefetch_b8 next K-tile
      __builtin_prefetch(bp + 32, 0, 3);
    }
    __syncthreads();
    v16bf af[2];
    #pragma unroll
    for(int mt=0; mt<2; mt++){
      int mrow = wm*32 + mt*16 + lcol;
      #pragma unroll
      for(int v=0; v<8; v++){
        int kk = halfl*16 + 2*v;
        af[mt][2*v]   = As[mrow][kk];
        af[mt][2*v+1] = As[mrow][kk+1];
      }
    }
    #pragma unroll
    for(int nt=0; nt<4; nt++){
      int ncol = wn*64 + nt*16 + lcol;
      v16bf bfr;
      #pragma unroll
      for(int v=0; v<8; v++){
        int kk = halfl*16 + 2*v;            // 16 contiguous K per lane
        bfr[2*v]   = BsT[ncol][kk];
        bfr[2*v+1] = BsT[ncol][kk+1];
      }
      #pragma unroll
      for(int mt=0; mt<2; mt++){
        acc[mt][nt] = __builtin_amdgcn_wmma_f32_16x16x32_bf16(
            false, af[mt], false, bfr, (short)0, acc[mt][nt], false, false);
      }
    }
    __syncthreads();
  }
  #pragma unroll
  for(int mt=0; mt<2; mt++){
    #pragma unroll
    for(int nt=0; nt<4; nt++){
      #pragma unroll
      for(int r=0; r<8; r++){
        int row = m0 + wm*32 + mt*16 + halfl*8 + r;
        int col = n0 + wn*64 + nt*16 + lcol;
        float v = acc[mt][nt][r];
        if(bias) v += bias[col];
        if(act == 1) v = v * sigmf(v);
        if(res) v += res[(size_t)row*N + col];
        C[(size_t)row*N + col] = v;
      }
    }
  }
}

// -------------------- GLU --------------------
__global__ void glu_kernel(const float* __restrict__ in, float* __restrict__ out){
  size_t n = (size_t)MTOK*CINNER;
  for(size_t i = (size_t)blockIdx.x*blockDim.x + threadIdx.x; i < n;
      i += (size_t)gridDim.x*blockDim.x){
    size_t m = i >> 10; int c = (int)(i & 1023);
    float a = in[m*2048 + c], g = in[m*2048 + 1024 + c];
    out[i] = a * sigmf(g);
  }
}

// ------------- causal depthwise conv + swish + channel LN -> bf16 -----------
__global__ __launch_bounds__(256) void dwconv_kernel(const float* __restrict__ in,
    const float* __restrict__ w, const float* __restrict__ b,
    const float* __restrict__ g, __bf16* __restrict__ out){
  __shared__ float rsum[256], rsq[256];
  int m = blockIdx.x; int bb = m >> 13; int t = m & (SEQL-1);
  int tid = threadIdx.x;
  float h[4];
  #pragma unroll
  for(int jj=0;jj<4;jj++){
    int c = tid + jj*256;
    float a = b[c];
    for(int k=0;k<KSZ;k++){
      int ts = t - (KSZ-1) + k;
      if(ts >= 0) a += in[(size_t)(bb*SEQL + ts)*CINNER + c] * w[k*CINNER + c];
    }
    h[jj] = a * sigmf(a);
  }
  float s  = h[0]+h[1]+h[2]+h[3];
  float s2 = h[0]*h[0]+h[1]*h[1]+h[2]*h[2]+h[3]*h[3];
  rsum[tid]=s; rsq[tid]=s2; __syncthreads();
  for(int st=128; st>0; st>>=1){
    if(tid<st){ rsum[tid]+=rsum[tid+st]; rsq[tid]+=rsq[tid+st]; }
    __syncthreads();
  }
  float mean = rsum[0]*(1.f/CINNER);
  float var  = rsq[0]*(1.f/CINNER) - mean*mean;
  float rs = rsqrtf(fmaxf(var, 1e-6f));
  #pragma unroll
  for(int jj=0;jj<4;jj++){
    int c = tid + jj*256;
    out[(size_t)m*CINNER + c] = (__bf16)((h[jj]-mean)*rs*g[c]);
  }
}

// -------------------- flash local attention --------------------
// one block per (batch, head, window); 256 thr = 8 waves; wave owns 16 q rows.
__global__ __launch_bounds__(256) void attn_kernel(const float* __restrict__ qkv,
    const float* __restrict__ bias, const float* __restrict__ qscale,
    const float* __restrict__ kscale, float* __restrict__ ao){
  __shared__ __align__(16) __bf16 Qs[128][64];   // 16 KB  [qrow][d]
  __shared__ __align__(16) __bf16 Ks[64][64];    //  8 KB  [key][d]
  __shared__ __align__(16) __bf16 VsT[64][64];   //  8 KB  [d][key]  (transposed!)
  __shared__ __align__(16) __bf16 Ps[8][16][64]; // 16 KB  [wave][qrow][key]
  int blk = blockIdx.x;
  int x = blk & (NWIN-1); int h = (blk >> 6) & (NHEADS-1); int bb = blk >> 9;
  int tid = threadIdx.x;
  int w_id = tid >> 5, lane = tid & 31;
  int halfl = lane >> 4, lcol = lane & 15;
  int ibase = w_id * 16;

  // ---- preprocess Q: l2norm * qscale, xpos rotary (*sc), *QK_SCALE ----
  if(tid < 128){
    int i = tid; int t = x*WINL + i;
    const float* qp = qkv + ((size_t)(bb*SEQL + t))*1536 + h*DHEAD;
    float q[64]; float ss = 0.f;
    #pragma unroll
    for(int d=0; d<64; d++){ q[d] = qp[d]; ss += q[d]*q[d]; }
    float invn = 1.f / fmaxf(sqrtf(ss), 1e-12f);
    #pragma unroll
    for(int d=0; d<64; d++) q[d] *= invn * qscale[d];
    float power = ((float)t - (float)(SEQL/2)) * (1.f/64.f);
    #pragma unroll
    for(int dd=0; dd<32; dd++){
      float fr = __powf(10000.f, -(2.f*dd)*(1.f/64.f));
      float sn, cs; __sincosf((float)t * fr, &sn, &cs);
      float sv = (2.f*dd + 25.6f) * (1.f/89.6f);
      float sc = __powf(sv, power);
      float x0 = q[dd], x1 = q[dd+32];
      q[dd]    = (x0*cs - x1*sn) * sc * 8.f;
      q[dd+32] = (x1*cs + x0*sn) * sc * 8.f;
    }
    #pragma unroll
    for(int d=0; d<64; d++) Qs[i][d] = (__bf16)q[d];
  }

  v8f acc[4] = {};
  float mrow[8], lrow[8];
  #pragma unroll
  for(int r=0;r<8;r++){ mrow[r] = -1e30f; lrow[r] = 0.f; }

  for(int c=0; c<4; c++){
    __syncthreads();  // protect previous iteration's LDS reads
    // ---- K chunk: l2norm * kscale, rotary (/sc); V chunk transposed ----
    if(tid < 64){
      int jj = tid; int kt = x*WINL - WINL + c*64 + jj;
      float k[64];
      if(kt >= 0){
        const float* kp = qkv + ((size_t)(bb*SEQL + kt))*1536 + 512 + h*DHEAD;
        float ss = 0.f;
        #pragma unroll
        for(int d=0; d<64; d++){ k[d] = kp[d]; ss += k[d]*k[d]; }
        float invn = 1.f / fmaxf(sqrtf(ss), 1e-12f);
        #pragma unroll
        for(int d=0; d<64; d++) k[d] *= invn * kscale[d];
        float power = ((float)kt - (float)(SEQL/2)) * (1.f/64.f);
        #pragma unroll
        for(int dd=0; dd<32; dd++){
          float fr = __powf(10000.f, -(2.f*dd)*(1.f/64.f));
          float sn, cs; __sincosf((float)kt * fr, &sn, &cs);
          float sv = (2.f*dd + 25.6f) * (1.f/89.6f);
          float isc = __powf(sv, -power);
          float x0 = k[dd], x1 = k[dd+32];
          k[dd]    = (x0*cs - x1*sn) * isc;
          k[dd+32] = (x1*cs + x0*sn) * isc;
        }
      } else {
        #pragma unroll
        for(int d=0; d<64; d++) k[d] = 0.f;
      }
      #pragma unroll
      for(int d=0; d<64; d++) Ks[jj][d] = (__bf16)k[d];
    } else if(tid < 128){
      int jj = tid - 64; int kt = x*WINL - WINL + c*64 + jj;
      const float* vp = qkv + ((size_t)(bb*SEQL + (kt<0?0:kt)))*1536 + 1024 + h*DHEAD;
      #pragma unroll
      for(int d=0; d<64; d++) VsT[d][jj] = (kt < 0) ? (__bf16)0.f : (__bf16)vp[d];
    }
    __syncthreads();

    // ---- S = Q @ K^T (16 rows x 64 cols per wave) ----
    v16bf aq[2];
    #pragma unroll
    for(int ks=0; ks<2; ks++){
      #pragma unroll
      for(int v=0; v<8; v++){
        int kk = ks*32 + halfl*16 + 2*v;
        aq[ks][2*v]   = Qs[ibase + lcol][kk];
        aq[ks][2*v+1] = Qs[ibase + lcol][kk+1];
      }
    }
    v8f sv4[4];
    #pragma unroll
    for(int nt=0; nt<4; nt++){
      v8f s = {};
      #pragma unroll
      for(int ks=0; ks<2; ks++){
        v16bf bk;
        #pragma unroll
        for(int v=0; v<8; v++){
          int kk = ks*32 + halfl*16 + 2*v;       // contiguous along d
          bk[2*v]   = Ks[nt*16 + lcol][kk];
          bk[2*v+1] = Ks[nt*16 + lcol][kk+1];
        }
        s = __builtin_amdgcn_wmma_f32_16x16x32_bf16(false, aq[ks], false, bk, (short)0, s, false, false);
      }
      sv4[nt] = s;
    }
    // ---- bias + causal window mask ----
    #pragma unroll
    for(int nt=0; nt<4; nt++){
      #pragma unroll
      for(int r=0; r<8; r++){
        int row = ibase + halfl*8 + r;
        int j = c*64 + nt*16 + lcol;
        int kt = x*WINL - WINL + j;
        bool ok = (kt >= 0) && (j >= row) && (j <= row + WINL);
        float bval = bias[(size_t)h*WINL*2*WINL + row*(2*WINL) + j];
        sv4[nt][r] = ok ? (sv4[nt][r] + bval) : -1e30f;
      }
    }
    // ---- online softmax ----
    #pragma unroll
    for(int r=0; r<8; r++){
      float ml = sv4[0][r];
      #pragma unroll
      for(int nt=1; nt<4; nt++) ml = fmaxf(ml, sv4[nt][r]);
      #pragma unroll
      for(int off=8; off>=1; off>>=1) ml = fmaxf(ml, __shfl_xor(ml, off, 32));
      float mnew = fmaxf(mrow[r], ml);
      float scal = __expf(mrow[r] - mnew);
      lrow[r] *= scal;
      #pragma unroll
      for(int nt=0; nt<4; nt++) acc[nt][r] *= scal;
      mrow[r] = mnew;
      float ls = 0.f;
      #pragma unroll
      for(int nt=0; nt<4; nt++){
        float pv = (sv4[nt][r] <= -1e29f) ? 0.f : __expf(sv4[nt][r] - mnew);
        sv4[nt][r] = pv; ls += pv;
      }
      #pragma unroll
      for(int off=8; off>=1; off>>=1) ls += __shfl_xor(ls, off, 32);
      lrow[r] += ls;
    }
    // ---- stage P (C-layout -> LDS) ----
    #pragma unroll
    for(int nt=0; nt<4; nt++)
      #pragma unroll
      for(int r=0; r<8; r++)
        Ps[w_id][halfl*8 + r][nt*16 + lcol] = (__bf16)sv4[nt][r];
    __syncthreads();
    // ---- O += P @ V ----
    v16bf ap[2];
    #pragma unroll
    for(int ks=0; ks<2; ks++){
      #pragma unroll
      for(int v=0; v<8; v++){
        int kk = ks*32 + halfl*16 + 2*v;
        ap[ks][2*v]   = Ps[w_id][lcol][kk];
        ap[ks][2*v+1] = Ps[w_id][lcol][kk+1];
      }
    }
    #pragma unroll
    for(int nt=0; nt<4; nt++){
      #pragma unroll
      for(int ks=0; ks<2; ks++){
        v16bf bv;
        #pragma unroll
        for(int v=0; v<8; v++){
          int kk = ks*32 + halfl*16 + 2*v;       // contiguous along key in VsT
          bv[2*v]   = VsT[nt*16 + lcol][kk];
          bv[2*v+1] = VsT[nt*16 + lcol][kk+1];
        }
        acc[nt] = __builtin_amdgcn_wmma_f32_16x16x32_bf16(false, ap[ks], false, bv, (short)0, acc[nt], false, false);
      }
    }
  }
  // ---- normalize and write out (b,n,h,d) ----
  #pragma unroll
  for(int nt=0; nt<4; nt++){
    #pragma unroll
    for(int r=0; r<8; r++){
      int row = ibase + halfl*8 + r;
      size_t m = (size_t)bb*SEQL + x*WINL + row;
      float invl = 1.f / fmaxf(lrow[r], 1e-20f);
      ao[m*DIMM + h*DHEAD + nt*16 + lcol] = acc[nt][r] * invl;
    }
  }
}

// ====================== host orchestration ======================
static void cvt_launch(const float* in, __bf16* out, size_t n, hipStream_t s){
  cvt_kernel<<<1024, 256, 0, s>>>(in, out, n);
}
static void cvt_t_launch(const float* in, __bf16* out, int K, int N, hipStream_t s){
  cvt_t_kernel<<<dim3(N/32, K/32), 256, 0, s>>>(in, out, K, N);
}

extern "C" void kernel_launch(void* const* d_in, const int* in_sizes, int n_in,
                              void* d_out, int out_size, void* d_ws, size_t ws_size,
                              hipStream_t stream){
  (void)in_sizes; (void)n_in; (void)out_size; (void)ws_size;
  const int*   tokens   = (const int*)  d_in[0];
  const float* tok_emb  = (const float*)d_in[1];
  const float* pos_emb  = (const float*)d_in[2];
  const float* pb_w1    = (const float*)d_in[3];
  const float* pb_b1    = (const float*)d_in[4];
  const float* pb_w2    = (const float*)d_in[5];
  const float* pb_b2    = (const float*)d_in[6];
  const float* pb_w3    = (const float*)d_in[7];
  const float* pb_b3    = (const float*)d_in[8];
  const float* attn_ln_g= (const float*)d_in[9];
  const float* attn_ln_b= (const float*)d_in[10];
  const float* w_qkv    = (const float*)d_in[11];
  const float* q_scale  = (const float*)d_in[12];
  const float* k_scale  = (const float*)d_in[13];
  const float* w_out    = (const float*)d_in[14];
  const float* conv_ln_g= (const float*)d_in[15];
  const float* conv_ln_b= (const float*)d_in[16];
  const float* pw1_w    = (const float*)d_in[17];
  const float* pw1_b    = (const float*)d_in[18];
  const float* dw_w     = (const float*)d_in[19];
  const float* dw_b     = (const float*)d_in[20];
  const float* chan_ln_g= (const float*)d_in[21];
  const float* pw2_w    = (const float*)d_in[22];
  const float* pw2_b    = (const float*)d_in[23];
  const float* ff_ln_g  = (const float*)d_in[24];
  const float* ff_ln_b  = (const float*)d_in[25];
  const float* ff_w1    = (const float*)d_in[26];
  const float* ff_b1    = (const float*)d_in[27];
  const float* ff_w2    = (const float*)d_in[28];
  const float* ff_b2    = (const float*)d_in[29];
  const float* fin_g    = (const float*)d_in[30];
  const float* fin_b    = (const float*)d_in[31];
  const float* w_logits = (const float*)d_in[32];

  char* ws = (char*)d_ws;
  size_t off = 0;
  float* X    = (float*)(ws + off); off += (size_t)MTOK*DIMM*4;
  float* QKV  = (float*)(ws + off); off += (size_t)MTOK*1536*4;
  float* T1   = (float*)(ws + off); off += (size_t)MTOK*2048*4;
  float* T2   = (float*)(ws + off); off += (size_t)MTOK*1024*4;
  float* AO   = (float*)(ws + off); off += (size_t)MTOK*DIMM*4;
  float* BIAS = (float*)(ws + off); off += (size_t)NHEADS*WINL*2*WINL*4;
  __bf16* LNB = (__bf16*)(ws + off); off += (size_t)MTOK*DIMM*2;
  __bf16* BF1 = (__bf16*)(ws + off); off += (size_t)MTOK*2048*2;
  __bf16* WB  = (__bf16*)(ws + off);

  // bf16 transposed-weight layout [N,K]
  const size_t SZ_QKV = 512*1536, SZ_OUT = 512*512, SZ_PW1 = 512*2048,
               SZ_PW2 = 1024*512, SZ_FF1 = 512*2048, SZ_FF2 = 2048*512,
               SZ_LOG = 512*1024;
  const size_t LS = SZ_QKV + SZ_OUT + SZ_PW1 + SZ_PW2 + SZ_FF1 + SZ_FF2;
  __bf16 *WQKV[2], *WOUT[2], *WPW1[2], *WPW2[2], *WFF1[2], *WFF2[2];
  for(int l=0; l<2; l++){
    size_t wo = (size_t)l*LS;
    WQKV[l] = WB + wo;                 wo += SZ_QKV;
    WOUT[l] = WB + wo;                 wo += SZ_OUT;
    WPW1[l] = WB + wo;                 wo += SZ_PW1;
    WPW2[l] = WB + wo;                 wo += SZ_PW2;
    WFF1[l] = WB + wo;                 wo += SZ_FF1;
    WFF2[l] = WB + wo;
  }
  __bf16* WLOG = WB + 2*LS;

  // weight conversions with transpose (fp32 [K,N] -> bf16 [N,K]), once per launch
  for(int l=0; l<2; l++){
    cvt_t_launch(w_qkv + (size_t)l*SZ_QKV, WQKV[l], 512, 1536, stream);
    cvt_t_launch(w_out + (size_t)l*SZ_OUT, WOUT[l], 512, 512, stream);
    cvt_t_launch(pw1_w + (size_t)l*SZ_PW1, WPW1[l], 512, 2048, stream);
    cvt_t_launch(pw2_w + (size_t)l*SZ_PW2, WPW2[l], 1024, 512, stream);
    cvt_t_launch(ff_w1 + (size_t)l*SZ_FF1, WFF1[l], 512, 2048, stream);
    cvt_t_launch(ff_w2 + (size_t)l*SZ_FF2, WFF2[l], 2048, 512, stream);
  }
  cvt_t_launch(w_logits, WLOG, 512, 1024, stream);

  embed_kernel<<<8192, 256, 0, stream>>>(tokens, tok_emb, pos_emb, X);
  posbias_kernel<<<1, 256, 0, stream>>>(pb_w1, pb_b1, pb_w2, pb_b2, pb_w3, pb_b3, BIAS);

  const int MB = MTOK/128;  // 128 block rows
  for(int l=0; l<2; l++){
    // ---- local attention ----
    ln_kernel<<<MTOK/8, 256, 0, stream>>>(X, attn_ln_g + l*DIMM, attn_ln_b + l*DIMM, LNB);
    gemm_kernel<<<dim3(1536/128, MB), 256, 0, stream>>>(LNB, WQKV[l], QKV,
        nullptr, nullptr, MTOK, 1536, 512, 0);
    attn_kernel<<<NBATCH*NHEADS*NWIN, 256, 0, stream>>>(QKV, BIAS,
        q_scale + l*DHEAD, k_scale + l*DHEAD, AO);
    cvt_launch(AO, BF1, (size_t)MTOK*DIMM, stream);
    gemm_kernel<<<dim3(512/128, MB), 256, 0, stream>>>(BF1, WOUT[l], X,
        nullptr, X, MTOK, 512, 512, 0);
    // ---- conformer conv ----
    ln_kernel<<<MTOK/8, 256, 0, stream>>>(X, conv_ln_g + l*DIMM, conv_ln_b + l*DIMM, LNB);
    gemm_kernel<<<dim3(2048/128, MB), 256, 0, stream>>>(LNB, WPW1[l], T1,
        pw1_b + (size_t)l*2048, nullptr, MTOK, 2048, 512, 0);
    glu_kernel<<<8192, 256, 0, stream>>>(T1, T2);
    dwconv_kernel<<<MTOK, 256, 0, stream>>>(T2, dw_w + (size_t)l*KSZ*CINNER,
        dw_b + (size_t)l*CINNER, chan_ln_g + (size_t)l*CINNER, BF1);
    gemm_kernel<<<dim3(512/128, MB), 256, 0, stream>>>(BF1, WPW2[l], X,
        pw2_b + (size_t)l*512, X, MTOK, 512, 1024, 0);
    // ---- feedforward ----
    ln_kernel<<<MTOK/8, 256, 0, stream>>>(X, ff_ln_g + l*DIMM, ff_ln_b + l*DIMM, LNB);
    gemm_kernel<<<dim3(2048/128, MB), 256, 0, stream>>>(LNB, WFF1[l], T1,
        ff_b1 + (size_t)l*2048, nullptr, MTOK, 2048, 512, 1 /*swish*/);
    cvt_launch(T1, BF1, (size_t)MTOK*2048, stream);
    gemm_kernel<<<dim3(512/128, MB), 256, 0, stream>>>(BF1, WFF2[l], X,
        ff_b2 + (size_t)l*512, X, MTOK, 512, 2048, 0);
  }
  // ---- final LN + logits ----
  ln_kernel<<<MTOK/8, 256, 0, stream>>>(X, fin_g, fin_b, LNB);
  gemm_kernel<<<dim3(1024/128, MB), 256, 0, stream>>>(LNB, WLOG, (float*)d_out,
      nullptr, nullptr, MTOK, 1024, 512, 0);
}
